// LieCNP_17952963297845
// MI455X (gfx1250) — matrix operand order repaired and checked by reading, not compiled, not run on hardware
//
#include <hip/hip_runtime.h>
#include <hip/hip_bf16.h>
#include <stdint.h>

// ---------------------------------------------------------------------------
// LieCNP forward for MI455X (gfx1250, wave32).
// Pipeline: k_h (psi RBF smoothing) -> 4x { k_topk (threefry noise + top-25),
// k_lieconv<CI,CO,SWISH> (WMMA MLP + einsum + linear) } -> k_zero -> k_out.
// ---------------------------------------------------------------------------

typedef __attribute__((ext_vector_type(16))) _Float16 v16h;
typedef __attribute__((ext_vector_type(8)))  float    v8f;
typedef __attribute__((ext_vector_type(8)))  uint32_t v8u;
typedef __attribute__((ext_vector_type(4)))  float    v4f;   // for NT stores

#define GRID_N   28
#define NPTS     784                    // 28*28
#define BATCH    32
#define KNB      25
#define NPOINTS  (BATCH * NPTS)         // 25088
#define NOISE_S  (BATCH * NPTS * NPTS)  // 19668992
#define NOISE_H  (NOISE_S / 2)
#define STEP     (28.0f / 27.0f)
#define CLIM     14.0f

// ---------------- threefry2x32 (JAX-compatible, classic counter scheme) ----
__host__ __device__ inline void threefry2x32(uint32_t k0, uint32_t k1,
                                             uint32_t c0, uint32_t c1,
                                             uint32_t& o0, uint32_t& o1) {
  const uint32_t rotA[4] = {13, 15, 26, 6};
  const uint32_t rotB[4] = {17, 29, 16, 24};
  uint32_t ks[3] = {k0, k1, k0 ^ k1 ^ 0x1BD11BDAu};
  uint32_t x0 = c0 + ks[0], x1 = c1 + ks[1];
#pragma unroll
  for (int i = 0; i < 5; ++i) {
    const uint32_t* r = (i & 1) ? rotB : rotA;
#pragma unroll
    for (int j = 0; j < 4; ++j) {
      x0 += x1;
      x1 = (x1 << r[j]) | (x1 >> (32u - r[j]));
      x1 ^= x0;
    }
    x0 += ks[(i + 1) % 3];
    x1 += ks[(i + 2) % 3] + (uint32_t)(i + 1);
  }
  o0 = x0; o1 = x1;
}

// ---------------- device helpers ------------------------------------------
__device__ inline float gcoord(int t) { return -CLIM + (float)t * STEP; }

// swish via raw v_rcp_f32 (no IEEE div refinement chain).
__device__ inline float swishf(float x) {
  return x * __builtin_amdgcn_rcpf(1.0f + __expf(-x));
}

// Wave-local LDS fence: DS ops are in-order per wave; the only hazard is the
// compiler reordering ds_read before ds_write across cross-lane handoffs.
__device__ inline void wave_lds_fence() {
  __builtin_amdgcn_wave_barrier();
  asm volatile("s_wait_dscnt 0x0" ::: "memory");
  __builtin_amdgcn_wave_barrier();
}

__device__ inline v8f wmma_f16(v16h a, v16h b, v8f c) {
  // v_wmma_f32_16x16x32_f16: D(16x16,f32) = A(16x32,f16) * B(32x16,f16) + C
  return __builtin_amdgcn_wmma_f32_16x16x32_f16(false, a, false, b,
                                                (short)0, c, false, false);
}

// A fragment (16x32 f16): lanes 0-15 -> M=lane, K 0..15;
// lanes 16-31 -> M=lane-16, K 16..31. src row-major [M][ld].
// K-pairs are contiguous -> 8 dword loads (mergeable to 2x ds_load_b128).
__device__ inline v16h load_a_frag(const _Float16* src, int ld, int mtile, int lane) {
  int m  = mtile * 16 + (lane & 15);
  int kb = (lane < 16) ? 0 : 16;
  const uint32_t* p = (const uint32_t*)(src + m * ld);
  v8u a;
#pragma unroll
  for (int v = 0; v < 8; ++v) {
    int k0 = kb + 2 * v;
    a[v] = p[k0 >> 1];
  }
  return __builtin_bit_cast(v16h, a);
}

// Packed B fragment image: frag[l*16+h] = B[kb(l)+h][n(l)] with
// n = l&15 (+ col offset at pack time), kb = (l<16)?0:16. Per-lane data is 16
// contiguous halves = 32B -> 2x ds_load_b128.
__device__ inline v16h load_b_frag_packed(const _Float16* frag, int lane) {
  const uint32_t* p = (const uint32_t*)(frag + lane * 16);
  v8u b;
#pragma unroll
  for (int v = 0; v < 8; ++v) b[v] = p[v];
  return __builtin_bit_cast(v16h, b);
}

// D fragment (16x16 f32): VGPR i -> row i (lanes 0-15) / row 8+i (lanes 16-31),
// col = lane&15. Add bias, swish, store f16 to row-major LDS tile (A staging).
__device__ inline void store_d_swish(v8f d, const float* bias, _Float16* dst,
                                     int ld, int mt, int nt, int lane) {
  int nl = nt * 16 + (lane & 15);
  int mb = mt * 16 + ((lane < 16) ? 0 : 8);
  float bv = bias[nl];
#pragma unroll
  for (int i = 0; i < 8; ++i) {
    float x = d[i] + bv;
    dst[(mb + i) * ld + nl] = (_Float16)swishf(x);
  }
}

// Same, but write directly into packed-B-fragment layout (for the einsum's
// B operand KW): element (row,n) -> frag[(row<16 ? n : 16+n)*16 + (row&15)].
__device__ inline void store_d_swish_bfrag(v8f d, const float* bias,
                                           _Float16* frag, int mt, int lane) {
  int n  = lane & 15;
  int mb = mt * 16 + ((lane < 16) ? 0 : 8);
  float bv = bias[n];
#pragma unroll
  for (int i = 0; i < 8; ++i) {
    int row = mb + i;
    float x = d[i] + bv;
    int l = (row < 16) ? n : (16 + n);
    frag[l * 16 + (row & 15)] = (_Float16)swishf(x);
  }
}

// GVS-mode async global->LDS copy of 16 bytes (ASYNCcnt-tracked).
__device__ inline void async_g2lds_b128(uint32_t ldsaddr, uint32_t voff,
                                        const void* base) {
  asm volatile("global_load_async_to_lds_b128 %0, %1, %2 offset:0"
               :: "v"(ldsaddr), "v"(voff), "s"(base) : "memory");
}
__device__ inline void wait_asynccnt0() {
  asm volatile("s_wait_asynccnt 0x0" ::: "memory");
}

// ---------------- kernel 1: psi RBF smoothing ------------------------------
// vals0[b,n,0:4] = { gx, gy, h0, h1/(h0+1e-8) }, h = K_psi(rep, ctx) @ [1, v]
__global__ __launch_bounds__(256)
void k_h(const float* __restrict__ cc, const float* __restrict__ cv,
         const float* __restrict__ psi_ls_p, const float* __restrict__ psi_os_p,
         float* __restrict__ v0) {
  int b   = blockIdx.x >> 2;
  int n   = (blockIdx.x & 3) * 256 + threadIdx.x;
  __shared__ float sx[256], sy[256], sv[256];
  float ls = psi_ls_p[0], os = psi_os_p[0];
  float inv = 0.5f / (ls * ls);
  float gx = 0.f, gy = 0.f;
  if (n < NPTS) { gx = gcoord(n / GRID_N); gy = gcoord(n % GRID_N); }
  float s0 = 0.f, s1 = 0.f;
  for (int m0 = 0; m0 < NPTS; m0 += 256) {
    int m = m0 + threadIdx.x;
    if (m < NPTS) {
      sx[threadIdx.x] = cc[(b * NPTS + m) * 2 + 0];
      sy[threadIdx.x] = cc[(b * NPTS + m) * 2 + 1];
      sv[threadIdx.x] = cv[b * NPTS + m];
    }
    __syncthreads();
    int lim = NPTS - m0; if (lim > 256) lim = 256;
    if (n < NPTS) {
      for (int t = 0; t < lim; ++t) {
        float dx = gx - sx[t], dy = gy - sy[t];
        float k = os * __expf(-(dx * dx + dy * dy) * inv);
        s0 += k; s1 += k * sv[t];
      }
    }
    __syncthreads();
  }
  if (n < NPTS) {
    float* o = v0 + (size_t)(b * NPTS + n) * 4;
    o[0] = gx; o[1] = gy; o[2] = s0; o[3] = s1 / (s0 + 1e-8f);
  }
}

// ---------------- kernel 2: threefry noise + top-25 (wave per row) ---------
__global__ __launch_bounds__(128)
void k_topk(uint32_t k0, uint32_t k1, int* __restrict__ idxOut) {
  int wave = threadIdx.x >> 5, lane = threadIdx.x & 31;
  int row  = blockIdx.x * 4 + wave;              // b*784 + n
  __shared__ float srow[4][NPTS];
  if (row < NPOINTS) {
    uint32_t gbase = (uint32_t)row * NPTS;
    for (int m = lane; m < NPTS; m += 32) {
      uint32_t g = gbase + (uint32_t)m;
      uint32_t c0, c1, r0, r1;
      bool lo = g < (uint32_t)NOISE_H;
      if (lo) { c0 = g; c1 = g + NOISE_H; } else { c0 = g - NOISE_H; c1 = g; }
      threefry2x32(k0, k1, c0, c1, r0, r1);
      uint32_t bits = lo ? r0 : r1;
      srow[wave][m] = __uint_as_float((bits >> 9) | 0x3f800000u) - 1.0f;
    }
  }
  wave_lds_fence();
  if (row >= NPOINTS) return;
  for (int t = 0; t < KNB; ++t) {
    float best = -1.0f; int bi = 0;
    for (int m = lane; m < NPTS; m += 32) {
      float v = srow[wave][m];
      if (v > best) { best = v; bi = m; }
    }
#pragma unroll
    for (int off = 16; off > 0; off >>= 1) {
      float ov = __shfl_xor(best, off, 32);
      int   oi = __shfl_xor(bi,   off, 32);
      if (ov > best || (ov == best && oi < bi)) { best = ov; bi = oi; }
    }
    if (lane == 0) {
      idxOut[row * KNB + t] = bi;
      srow[wave][bi] = -1.0f;
    }
    wave_lds_fence();
  }
}

// ---------------- kernel 3: LieConv layer (wave per point, WMMA MLP) -------
template <int CI, int CO, int SWISH>
__global__ __launch_bounds__(128)
void k_lieconv(const int* __restrict__ idx,
               const float* __restrict__ vin, float* __restrict__ vout,
               const float* __restrict__ W1, const float* __restrict__ b1,
               const float* __restrict__ W2, const float* __restrict__ b2,
               const float* __restrict__ W3, const float* __restrict__ b3,
               const float* __restrict__ Wl, const float* __restrict__ bl) {
  const int tid = threadIdx.x, wave = tid >> 5, lane = tid & 31;
  __shared__ float    sW2raw[32 * 32];     // async-load landing zone (f32)
  __shared__ _Float16 sWB1[2][32 * 16];    // W1 packed B-frag tiles (n 0..15, 16..31)
  __shared__ _Float16 sWB2[2][32 * 16];    // W2 packed B-frag tiles
  __shared__ _Float16 sWB3[32 * 16];       // W3 packed B-frag
  __shared__ float    sB1[32], sB2[32], sB3[16], sBl[32];
  __shared__ _Float16 sA[4][32 * 32];      // A staging: rel / H1 / H2 / nbT
  __shared__ _Float16 sKWf[4][32 * 16];    // KW in packed B-frag layout
  __shared__ float    sP[4][32 * 16];      // partial (CI x 16) f32

  // Async DMA the 4KB W2 block straight into LDS (ASYNCcnt path), then
  // pack the small tensors with plain loads while the DMA is in flight.
  {
    uint32_t ldsbase = (uint32_t)(uintptr_t)&sW2raw[0];
#pragma unroll
    for (int it = 0; it < 2; ++it) {
      uint32_t off = (uint32_t)(tid + it * 128) * 16u;   // 256 x 16B = 4KB
      async_g2lds_b128(ldsbase + off, off, W2);
    }
  }
  // W1 (2x32, zero-pad K to 32) and W3 (32x16) packed directly from global.
  for (int e = tid; e < 512; e += 128) {
    int l = e >> 4, h = e & 15;
    int n = l & 15, k = ((l < 16) ? 0 : 16) + h;
    sWB1[0][e] = (_Float16)((k < 2) ? W1[k * 32 + n] : 0.0f);
    sWB1[1][e] = (_Float16)((k < 2) ? W1[k * 32 + n + 16] : 0.0f);
    sWB3[e]    = (_Float16)W3[k * 16 + n];
  }
  if (tid < 32) {
    sB1[tid] = b1[tid];
    sB2[tid] = b2[tid];
    sBl[tid] = (tid < CO) ? bl[tid] : 0.0f;
  }
  if (tid < 16) sB3[tid] = b3[tid];
  wait_asynccnt0();
  __syncthreads();
  for (int e = tid; e < 512; e += 128) {     // pack W2 tiles from raw f32
    int l = e >> 4, h = e & 15;
    int n = l & 15, k = ((l < 16) ? 0 : 16) + h;
    sWB2[0][e] = (_Float16)sW2raw[k * 32 + n];
    sWB2[1][e] = (_Float16)sW2raw[k * 32 + n + 16];
  }
  __syncthreads();

  int p = blockIdx.x * 4 + wave;
  if (p >= NPOINTS) return;
  int b = p / NPTS, n = p % NPTS;
  const int* myidx = idx + (size_t)p * KNB;
  __builtin_prefetch(Wl, 0, 1);   // global_prefetch for the final linear

  // --- stage rel = nb_coords - center (25x2, zero-padded to 32x32, f16) ----
  for (int e = lane; e < 1024; e += 32) sA[wave][e] = (_Float16)0.0f;
  wave_lds_fence();
  float cx = gcoord(n / GRID_N), cy = gcoord(n % GRID_N);
  for (int k = lane; k < KNB; k += 32) {
    int j = myidx[k];
    sA[wave][k * 32 + 0] = (_Float16)(gcoord(j / GRID_N) - cx);
    sA[wave][k * 32 + 1] = (_Float16)(gcoord(j % GRID_N) - cy);
  }
  wave_lds_fence();

  v8f z = {};
  // --- GEMM1: H1 = swish(rel @ W1 + b1)  (K padded 2->32) ------------------
  {
    v16h a0 = load_a_frag(sA[wave], 32, 0, lane);
    v16h a1 = load_a_frag(sA[wave], 32, 1, lane);
    v16h w0 = load_b_frag_packed(sWB1[0], lane);
    v16h w1 = load_b_frag_packed(sWB1[1], lane);
    v8f d00 = wmma_f16(a0, w0, z), d01 = wmma_f16(a0, w1, z);
    v8f d10 = wmma_f16(a1, w0, z), d11 = wmma_f16(a1, w1, z);
    wave_lds_fence();
    store_d_swish(d00, sB1, sA[wave], 32, 0, 0, lane);
    store_d_swish(d01, sB1, sA[wave], 32, 0, 1, lane);
    store_d_swish(d10, sB1, sA[wave], 32, 1, 0, lane);
    store_d_swish(d11, sB1, sA[wave], 32, 1, 1, lane);
  }
  wave_lds_fence();
  // --- GEMM2: H2 = swish(H1 @ W2 + b2)  (K=32 in one WMMA) -----------------
  {
    v16h a0 = load_a_frag(sA[wave], 32, 0, lane);
    v16h a1 = load_a_frag(sA[wave], 32, 1, lane);
    v16h w0 = load_b_frag_packed(sWB2[0], lane);
    v16h w1 = load_b_frag_packed(sWB2[1], lane);
    v8f d00 = wmma_f16(a0, w0, z), d01 = wmma_f16(a0, w1, z);
    v8f d10 = wmma_f16(a1, w0, z), d11 = wmma_f16(a1, w1, z);
    wave_lds_fence();
    store_d_swish(d00, sB2, sA[wave], 32, 0, 0, lane);
    store_d_swish(d01, sB2, sA[wave], 32, 0, 1, lane);
    store_d_swish(d10, sB2, sA[wave], 32, 1, 0, lane);
    store_d_swish(d11, sB2, sA[wave], 32, 1, 1, lane);
  }
  wave_lds_fence();
  // --- GEMM3: KW = swish(H2 @ W3 + b3) -> sKWf (packed B-frag) -------------
  {
    v16h a0 = load_a_frag(sA[wave], 32, 0, lane);
    v16h a1 = load_a_frag(sA[wave], 32, 1, lane);
    v16h w3 = load_b_frag_packed(sWB3, lane);
    v8f d0 = wmma_f16(a0, w3, z), d1 = wmma_f16(a1, w3, z);
    wave_lds_fence();
    store_d_swish_bfrag(d0, sB3, sKWf[wave], 0, lane);
    store_d_swish_bfrag(d1, sB3, sKWf[wave], 1, lane);
  }
  wave_lds_fence();
  // --- stage nb_vals^T (CI x 25, zero-padded K columns => exact padding) ---
  for (int e = lane; e < 1024; e += 32) sA[wave][e] = (_Float16)0.0f;
  wave_lds_fence();
  for (int e = lane; e < CI * KNB; e += 32) {
    int c = e / KNB, k = e % KNB;
    float v = vin[(size_t)(b * NPTS + myidx[k]) * CI + c];
    if (SWISH) v = swishf(v);
    sA[wave][c * 32 + k] = (_Float16)v;
  }
  wave_lds_fence();
  // --- GEMM4: partial(CI x 16) = nb_vals^T @ KW ----------------------------
  {
    v16h bkw = load_b_frag_packed(sKWf[wave], lane);
    constexpr int MT = (CI + 15) / 16;
#pragma unroll
    for (int mt = 0; mt < MT; ++mt) {
      v16h av = load_a_frag(sA[wave], 32, mt, lane);
      v8f dp = wmma_f16(av, bkw, z);
      int nl = lane & 15;
      int mb = mt * 16 + ((lane < 16) ? 0 : 8);
#pragma unroll
      for (int i = 0; i < 8; ++i) sP[wave][(mb + i) * 16 + nl] = dp[i];
    }
  }
  wave_lds_fence();
  // --- GEMM5: out = flatten(partial) @ Wl + bl (VALU dot, lane per output) -
  for (int j = lane; j < CO; j += 32) {
    float acc = sBl[j];
    for (int c = 0; c < CI; ++c)
#pragma unroll
      for (int d = 0; d < 16; ++d)
        acc += sP[wave][c * 16 + d] * Wl[(size_t)(c * 16 + d) * CO + j];
    vout[(size_t)(b * NPTS + n) * CO + j] = acc;
  }
}

// ---------------- kernel 4: zero-fill sigma (NT b128 stores) ---------------
__global__ __launch_bounds__(256)
void k_zero(v4f* __restrict__ p, long n4) {
  const v4f z = {0.f, 0.f, 0.f, 0.f};
  long stride = (long)gridDim.x * blockDim.x;
  for (long i = (long)blockIdx.x * blockDim.x + threadIdx.x; i < n4; i += stride)
    __builtin_nontemporal_store(z, &p[i]);   // streaming fill, spare L2
}

// ---------------- kernel 5: rho readout (mu + sigma diagonal) --------------
__global__ __launch_bounds__(256)
void k_out(const float* __restrict__ tc, const float* __restrict__ vfin,
           const float* __restrict__ rho_ls_p, const float* __restrict__ rho_os_p,
           float* __restrict__ out) {
  int b = blockIdx.x >> 2;
  int t = (blockIdx.x & 3) * 256 + threadIdx.x;
  __shared__ float fm[NPTS], fs[NPTS], fgx[NPTS], fgy[NPTS];
  for (int e = threadIdx.x; e < NPTS; e += 256) {
    float mu_ = vfin[(size_t)(b * NPTS + e) * 2 + 0];
    float sg_ = vfin[(size_t)(b * NPTS + e) * 2 + 1];
    fm[e] = mu_;
    fs[e] = (sg_ > 20.0f) ? sg_ : log1pf(__expf(sg_));   // softplus
    fgx[e] = gcoord(e / GRID_N);
    fgy[e] = gcoord(e % GRID_N);
  }
  __syncthreads();
  if (t >= NPTS) return;
  float tx = tc[(size_t)(b * NPTS + t) * 2 + 0];
  float ty = tc[(size_t)(b * NPTS + t) * 2 + 1];
  float ls = rho_ls_p[0], os = rho_os_p[0];
  float inv = 0.5f / (ls * ls);
  float mu = 0.f, sg = 0.f;
  for (int n2 = 0; n2 < NPTS; ++n2) {
    float dx = tx - fgx[n2], dy = ty - fgy[n2];
    float k = os * __expf(-(dx * dx + dy * dy) * inv);
    mu += k * fm[n2]; sg += k * fs[n2];
  }
  out[(size_t)b * NPTS + t] = mu;
  out[(size_t)NPOINTS + (size_t)b * NPTS * NPTS + (size_t)t * (NPTS + 1)] = sg;
}

// ---------------------------------------------------------------------------
extern "C" void kernel_launch(void* const* d_in, const int* in_sizes, int n_in,
                              void* d_out, int out_size, void* d_ws, size_t ws_size,
                              hipStream_t stream) {
  (void)in_sizes; (void)n_in; (void)out_size; (void)ws_size;
  const float* ctx_coords = (const float*)d_in[0];
  const float* ctx_values = (const float*)d_in[1];
  // d_in[2] = ctx_mask (all ones, unused by reference)
  const float* tgt_coords = (const float*)d_in[3];
  // params leaves, pytree order: layers[0..3]{W1,W2,W3,Wl,b1,b2,b3,bl}, then
  // psi_ls, psi_os, rho_ls, rho_os.
  const float *W1[4], *W2[4], *W3[4], *Wl[4], *b1[4], *b2[4], *b3[4], *bl[4];
  for (int li = 0; li < 4; ++li) {
    int base = 4 + li * 8;
    W1[li] = (const float*)d_in[base + 0];
    W2[li] = (const float*)d_in[base + 1];
    W3[li] = (const float*)d_in[base + 2];
    Wl[li] = (const float*)d_in[base + 3];
    b1[li] = (const float*)d_in[base + 4];
    b2[li] = (const float*)d_in[base + 5];
    b3[li] = (const float*)d_in[base + 6];
    bl[li] = (const float*)d_in[base + 7];
  }
  const float* psi_ls = (const float*)d_in[36];
  const float* psi_os = (const float*)d_in[37];
  const float* rho_ls = (const float*)d_in[38];
  const float* rho_os = (const float*)d_in[39];

  // Derive the 4 layer PRNG keys on host: split(key(42), 4).
  uint32_t bits[8];
  for (int j = 0; j < 4; ++j) {
    uint32_t a, c;
    threefry2x32(0u, 42u, (uint32_t)j, (uint32_t)(j + 4), a, c);
    bits[j] = a; bits[4 + j] = c;
  }
  uint32_t key0[4] = {bits[0], bits[2], bits[4], bits[6]};
  uint32_t key1[4] = {bits[1], bits[3], bits[5], bits[7]};

  // Workspace: two ping-pong value buffers (stride up to 32) + idx buffer.
  float* vA = (float*)d_ws;                       // NPOINTS*32 floats
  float* vB = vA + (size_t)NPOINTS * 32;          // NPOINTS*32 floats
  int*   idxbuf = (int*)(vB + (size_t)NPOINTS * 32);

  k_h<<<BATCH * 4, 256, 0, stream>>>(ctx_coords, ctx_values, psi_ls, psi_os, vA);

  const int NB = NPOINTS / 4;
  // Layer 0: ci=4, co=16, no input swish.
  k_topk<<<NB, 128, 0, stream>>>(key0[0], key1[0], idxbuf);
  k_lieconv<4, 16, 0><<<NB, 128, 0, stream>>>(idxbuf, vA, vB,
      W1[0], b1[0], W2[0], b2[0], W3[0], b3[0], Wl[0], bl[0]);
  // Layer 1: ci=16, co=32, swish.
  k_topk<<<NB, 128, 0, stream>>>(key0[1], key1[1], idxbuf);
  k_lieconv<16, 32, 1><<<NB, 128, 0, stream>>>(idxbuf, vB, vA,
      W1[1], b1[1], W2[1], b2[1], W3[1], b3[1], Wl[1], bl[1]);
  // Layer 2: ci=32, co=16, swish.
  k_topk<<<NB, 128, 0, stream>>>(key0[2], key1[2], idxbuf);
  k_lieconv<32, 16, 1><<<NB, 128, 0, stream>>>(idxbuf, vA, vB,
      W1[2], b1[2], W2[2], b2[2], W3[2], b3[2], Wl[2], bl[2]);
  // Layer 3: ci=16, co=2, swish.
  k_topk<<<NB, 128, 0, stream>>>(key0[3], key1[3], idxbuf);
  k_lieconv<16, 2, 1><<<NB, 128, 0, stream>>>(idxbuf, vB, vA,
      W1[3], b1[3], W2[3], b2[3], W3[3], b3[3], Wl[3], bl[3]);
  // vA now holds f_mu/f_sigma (stride 2).

  float* sigma_base = (float*)d_out + NPOINTS;    // 25088 % 4 == 0 -> b128 ok
  long n4 = (long)BATCH * NPTS * NPTS / 4;
  k_zero<<<2048, 256, 0, stream>>>((v4f*)sigma_base, n4);
  k_out<<<BATCH * 4, 256, 0, stream>>>(tgt_coords, vA, rho_ls, rho_os,
                                       (float*)d_out);
}
